// DynamicGPCM_58918361366957
// MI455X (gfx1250) — matrix operand order repaired and checked by reading, not compiled, not run on hardware
//
#include <hip/hip_runtime.h>
#include <stdint.h>

// Dynamic GPCM recurrent scan for MI455X (gfx1250, wave32, WMMA).
// Each wave owns a 16-student tile and runs all 512 steps; the three MLP
// GEMMs per step are done with v_wmma_f32_16x16x32_f16.

typedef __attribute__((ext_vector_type(16))) _Float16 v16h;
typedef __attribute__((ext_vector_type(8)))  float    v8f;

#define S_LEN  512
#define B_ROWS 4096

// output section offsets in floats (theta, alpha, beta, logits, probs)
#define OFF_THETA 0ull
#define OFF_ALPHA 33554432ull
#define OFF_BETA  67108864ull
#define OFF_LOGIT 75497472ull
#define OFF_PROB  85983232ull

union FragU { v16h h; uint32_t u[8]; };

static __device__ inline v8f wmma_f16(v16h a, v16h b, v8f c) {
    // D = A(16x32 f16) * B(32x16 f16) + C(16x16 f32)
    return __builtin_amdgcn_wmma_f32_16x16x32_f16(false, a, false, b, (short)0, c, false, false);
}

// Branch-free activations: single v_exp_f32 + v_rcp_f32, no exec-mask paths.
static __device__ inline float fast_tanh(float x) {
    // 1 - 2/(e^{2x}+1); exp->inf/0 saturate correctly to +/-1.
    float e = __expf(2.0f * x);
    return 1.0f - 2.0f * __builtin_amdgcn_rcpf(e + 1.0f);
}
static __device__ inline float fast_sigmoid(float x) {
    return __builtin_amdgcn_rcpf(1.0f + __expf(-x));
}

// A-matrix fragment: row-major LDS row (contiguous k), ISA layout:
// halves 0..7 -> K = mb8+0..7, halves 8..15 -> K = mb8+16..23 (mb8 = 0|8).
static __device__ inline v16h load_a_frag(const _Float16* row, int kbase, int mb8) {
    const uint32_t* p = (const uint32_t*)(row) + ((kbase + mb8) >> 1);
    FragU f;
#pragma unroll
    for (int j = 0; j < 4; ++j) { f.u[j] = p[j]; f.u[j + 4] = p[j + 8]; }
    return f.h;
}

// B-matrix fragment: transposed LDS row [n][k] (contiguous k), ISA layout:
// halves 0..7 -> K = kbase+kb8+0..7, halves 8..15 -> K = kbase+kb8+16..23 (kb8 = kb16/2).
static __device__ inline v16h load_b_frag(const _Float16* row, int kbase, int kb16) {
    const uint32_t* p = (const uint32_t*)(row) + ((kbase + (kb16 >> 1)) >> 1);
    FragU f;
#pragma unroll
    for (int j = 0; j < 4; ++j) { f.u[j] = p[j]; f.u[j + 4] = p[j + 8]; }
    return f.h;
}

// High K-chunk (k=32..63) of W_h: only k=32..36 nonzero (rows padded to 40).
// Unconditional in-bounds loads (halves 32..39), zeroed by a per-lane AND mask
// for lanes that represent K=40..47 and 56..63 — avoids exec-masked divergent ds loads.
static __device__ inline v16h load_whb_hi(const _Float16* row, uint32_t lmask) {
    const uint32_t* p = (const uint32_t*)(row) + 16;  // half offset 32
    FragU f;
#pragma unroll
    for (int j = 0; j < 4; ++j) f.u[j] = p[j] & lmask;
#pragma unroll
    for (int j = 4; j < 8; ++j) f.u[j] = 0u;
    return f.h;
}

__global__ __launch_bounds__(64)
void dynamic_gpcm_kernel(const float* __restrict__ theta_embed,
                         const float* __restrict__ alpha_raw,
                         const float* __restrict__ beta_base,
                         const float* __restrict__ beta_gaps,
                         const float* __restrict__ W_h,
                         const float* __restrict__ b_h,
                         const float* __restrict__ W_g,
                         const float* __restrict__ b_g,
                         const float* __restrict__ W_d,
                         const float* __restrict__ b_d,
                         const int*   __restrict__ student_ids,
                         const int*   __restrict__ questions,
                         const int*   __restrict__ responses,
                         float*       __restrict__ out)
{
    __shared__ __align__(16) _Float16 Wht_s[256 * 40];   // W_h^T [n][k], k padded to 40
    __shared__ __align__(16) _Float16 Wgt_s[16 * 256];   // W_g^T [n][k]
    __shared__ __align__(16) _Float16 Wdt_s[16 * 256];   // W_d^T [n][k]
    __shared__ float    bh_s[256];
    __shared__ float    bg_s[16];
    __shared__ float    bd_s[16];
    __shared__ __align__(16) float    theta_s[2][16 * 16];  // per-wave carry
    __shared__ __align__(16) _Float16 u_s[2][16 * 64];      // per-wave MLP input (zero-padded)
    __shared__ __align__(16) _Float16 h_s[2][16 * 256];     // per-wave hidden acts

    const int tid  = threadIdx.x;
    const int wid  = tid >> 5;
    const int lane = tid & 31;
    const int nn   = lane & 15;            // column / A-row index in 16-tile
    const int mb8  = (lane & 16) ? 8 : 0;  // A-frag & C/D row offset
    const int kb16 = (lane & 16) ? 16 : 0; // B-frag K offset
    const uint32_t hiMask = (lane & 16) ? 0u : 0xFFFFFFFFu; // W_h high-chunk lane mask
    const int rowBase = blockIdx.x * 32 + wid * 16;

    // ---- stage weights to LDS (block cooperative, once) ----
    for (int idx = tid; idx < 256 * 40; idx += 64) {
        int n = idx / 40, k = idx - n * 40;
        Wht_s[idx] = (_Float16)((k < 37) ? W_h[k * 256 + n] : 0.0f);
    }
    for (int idx = tid; idx < 16 * 256; idx += 64) {
        int n = idx >> 8, k = idx & 255;
        Wgt_s[idx] = (_Float16)W_g[k * 16 + n];
        Wdt_s[idx] = (_Float16)W_d[k * 16 + n];
    }
    for (int idx = tid; idx < 256; idx += 64) bh_s[idx] = b_h[idx];
    if (tid < 16) { bg_s[tid] = b_g[tid]; bd_s[tid] = b_d[tid]; }

    // per-wave init: zero u (padding cols persist as zero), load theta0
    for (int idx = lane; idx < 16 * 64; idx += 32) u_s[wid][idx] = (_Float16)0.0f;
    if (lane < 16) {
        int row = rowBase + lane;
        int sid = student_ids[row * S_LEN];
#pragma unroll
        for (int d = 0; d < 16; ++d)
            theta_s[wid][lane * 16 + d] = theta_embed[sid * 16 + d];
    }
    __syncthreads();

    const float bgv = bg_s[nn];
    const float bdv = bd_s[nn];

    for (int s = 0; s < S_LEN; ++s) {
        // ---- per-row GPCM stage (lanes 0..15, one student row each) ----
        if (lane < 16) {
            const int m   = lane;
            const int row = rowBase + m;
            const int q   = questions[row * S_LEN + s];
            const float rr = (float)responses[row * S_LEN + s];
            float th[16], al[16], z = 0.0f;
#pragma unroll
            for (int d = 0; d < 16; ++d) {
                th[d] = theta_s[wid][m * 16 + d];
                al[d] = __expf(0.3f * alpha_raw[q * 16 + d]);
                z += al[d] * th[d];
            }
            float be[4];
            {
                float acc = beta_base[q];
                be[0] = acc;
#pragma unroll
                for (int j = 0; j < 3; ++j) {
                    float x  = beta_gaps[q * 3 + j];
                    float sp = (x > 20.0f) ? x : __logf(1.0f + __expf(x));
                    acc += sp;
                    be[j + 1] = acc;
                }
            }
            float lg[5];
            lg[0] = 0.0f;
#pragma unroll
            for (int j = 1; j < 5; ++j) lg[j] = lg[j - 1] + z - be[j - 1];
            float mx = lg[0];
#pragma unroll
            for (int j = 1; j < 5; ++j) mx = fmaxf(mx, lg[j]);
            float pe[5], sum = 0.0f;
#pragma unroll
            for (int j = 0; j < 5; ++j) { pe[j] = __expf(lg[j] - mx); sum += pe[j]; }
            const float inv = __builtin_amdgcn_rcpf(sum);
            float expected = 0.0f;
#pragma unroll
            for (int j = 0; j < 5; ++j) { pe[j] *= inv; expected += (float)j * pe[j]; }
            const float sur = rr - expected;

            // ---- streamed outputs (theta is pre-update carry) ----
            const size_t pos = (size_t)row * S_LEN + s;
            float4* o_th = (float4*)(out + OFF_THETA + pos * 16);
            float4* o_al = (float4*)(out + OFF_ALPHA + pos * 16);
#pragma unroll
            for (int j4 = 0; j4 < 4; ++j4) {
                o_th[j4] = make_float4(th[4*j4], th[4*j4+1], th[4*j4+2], th[4*j4+3]);
                o_al[j4] = make_float4(al[4*j4], al[4*j4+1], al[4*j4+2], al[4*j4+3]);
            }
            *(float4*)(out + OFF_BETA + pos * 4) = make_float4(be[0], be[1], be[2], be[3]);
            float* o_lg = out + OFF_LOGIT + pos * 5;
            float* o_pr = out + OFF_PROB  + pos * 5;
#pragma unroll
            for (int j = 0; j < 5; ++j) { o_lg[j] = lg[j]; o_pr[j] = pe[j]; }

            // ---- build MLP input row u = [theta, surprise, alpha, beta] ----
            _Float16* ur = &u_s[wid][m * 64];
#pragma unroll
            for (int d = 0; d < 16; ++d) ur[d] = (_Float16)th[d];
            ur[16] = (_Float16)sur;
#pragma unroll
            for (int d = 0; d < 16; ++d) ur[17 + d] = (_Float16)al[d];
#pragma unroll
            for (int j = 0; j < 4; ++j) ur[33 + j] = (_Float16)be[j];
        }
        __syncthreads();

        // ---- GEMM1: h = tanh(u(16x64) @ W_h(64x256) + b_h) via WMMA ----
        {
            const _Float16* urow = &u_s[wid][nn * 64];
            v16h a0 = load_a_frag(urow, 0,  mb8);
            v16h a1 = load_a_frag(urow, 32, mb8);
#pragma unroll 4
            for (int nt = 0; nt < 16; ++nt) {
                const _Float16* wrow = &Wht_s[(nt * 16 + nn) * 40];
                v16h b0 = load_b_frag(wrow, 0, kb16);
                v16h b1 = load_whb_hi(wrow, hiMask);
                v8f c = {0.f,0.f,0.f,0.f,0.f,0.f,0.f,0.f};
                c = wmma_f16(a0, b0, c);
                c = wmma_f16(a1, b1, c);
                const float bias = bh_s[nt * 16 + nn];
#pragma unroll
                for (int v = 0; v < 8; ++v)
                    h_s[wid][(v + mb8) * 256 + nt * 16 + nn] =
                        (_Float16)fast_tanh(c[v] + bias);
            }
        }
        __syncthreads();

        // ---- GEMM2/3: gate & delta heads, h(16x256) @ W(256x16), then theta update ----
        {
            v8f cg = {0.f,0.f,0.f,0.f,0.f,0.f,0.f,0.f};
            v8f cd = {0.f,0.f,0.f,0.f,0.f,0.f,0.f,0.f};
            const _Float16* hrow = &h_s[wid][nn * 256];
            const _Float16* grow = &Wgt_s[nn * 256];
            const _Float16* drow = &Wdt_s[nn * 256];
#pragma unroll
            for (int kk = 0; kk < 8; ++kk) {
                v16h a  = load_a_frag(hrow, kk * 32, mb8);
                v16h bg = load_b_frag(grow, kk * 32, kb16);
                v16h bd = load_b_frag(drow, kk * 32, kb16);
                cg = wmma_f16(a, bg, cg);
                cd = wmma_f16(a, bd, cd);
            }
            // C/D tile element (v, lane) -> theta[v+mb8][nn]
#pragma unroll
            for (int v = 0; v < 8; ++v) {
                const float g  = fast_sigmoid(cg[v] + bgv);
                const float dl = fast_tanh(cd[v] + bdv);
                theta_s[wid][(v + mb8) * 16 + nn] += g * dl;
            }
        }
        __syncthreads();
    }
}

extern "C" void kernel_launch(void* const* d_in, const int* in_sizes, int n_in,
                              void* d_out, int out_size, void* d_ws, size_t ws_size,
                              hipStream_t stream) {
    const float* theta_embed = (const float*)d_in[0];
    const float* alpha_raw   = (const float*)d_in[1];
    const float* beta_base   = (const float*)d_in[2];
    const float* beta_gaps   = (const float*)d_in[3];
    const float* W_h         = (const float*)d_in[4];
    const float* b_h         = (const float*)d_in[5];
    const float* W_g         = (const float*)d_in[6];
    const float* b_g         = (const float*)d_in[7];
    const float* W_d         = (const float*)d_in[8];
    const float* b_d         = (const float*)d_in[9];
    const int*   student_ids = (const int*)d_in[10];
    const int*   questions   = (const int*)d_in[11];
    const int*   responses   = (const int*)d_in[12];
    float* out = (float*)d_out;

    dim3 grid(B_ROWS / 32);  // 128 blocks, 2 waves each, 16 student rows per wave
    dim3 block(64);
    dynamic_gpcm_kernel<<<grid, block, 0, stream>>>(
        theta_embed, alpha_raw, beta_base, beta_gaps,
        W_h, b_h, W_g, b_g, W_d, b_d,
        student_ids, questions, responses, out);
}
